// StaticPrunedViT_53532472377669
// MI455X (gfx1250) — compile-verified, matrix operand order checked by reading, not run on hardware
//
#include <hip/hip_runtime.h>
#include <math.h>

// ---------------- model constants ----------------
#define DEPTH  12
#define DIMV   768
#define HEADS  12
#define HDIM   64
#define MLPD   3072
#define NPATCH 196
#define NCLS   1000
#define KEEPT  128
#define NBATCH 32
#define NTOK1  197           // tokens pre-prune
#define NTOK2  129           // tokens post-prune (1 cls + 128)
#define SLD    208           // padded leading dim for attention score tiles

typedef __bf16 bf16;
typedef __attribute__((ext_vector_type(8)))  bf16  v8bf;
typedef __attribute__((ext_vector_type(16))) bf16  v16bf;
typedef __attribute__((ext_vector_type(8)))  float v8f;
typedef __attribute__((ext_vector_type(4)))  unsigned int v4u;
typedef __attribute__((ext_vector_type(8)))  int  v8i;
typedef __attribute__((ext_vector_type(4)))  int  v4i;

// Tensor Data Mover path available? (arity differs across toolchains)
#if defined(__has_builtin)
#if __has_builtin(__builtin_amdgcn_tensor_load_to_lds) && __has_builtin(__builtin_amdgcn_s_wait_tensorcnt)
#define GEMM_TDM 1
#endif
#endif
#ifndef GEMM_TDM
#define GEMM_TDM 0
#endif

__device__ __forceinline__ float gelu_exact(float v) {
    return 0.5f * v * (1.0f + erff(v * 0.70710678118654752f));
}

// load 32 bf16 (guarded, zero-filled) into 4 x v8bf  (non-TDM fallback)
__device__ __forceinline__ void load_chunk32(const bf16* __restrict__ src,
                                             bool row_ok, int krem, v8bf r[4]) {
    if (row_ok && krem >= 32) {
        const v8bf* s = (const v8bf*)src;
        r[0] = s[0]; r[1] = s[1]; r[2] = s[2]; r[3] = s[3];
    } else {
#pragma unroll
        for (int q = 0; q < 4; q++)
#pragma unroll
            for (int j = 0; j < 8; j++) {
                int kk = q * 8 + j;
                r[q][j] = (row_ok && kk < krem) ? src[kk] : (bf16)0.0f;
            }
    }
}

#if GEMM_TDM
// Issue one TDM 2D tile load: rows x 64 bf16 tile, global row stride `ld`
// elements, into LDS rows padded to 72 elements (pad 4 DWORDs per 32 DWORDs).
// tensor_dim0/1 = remaining extents from tile origin -> HW zero-fills OOB.
// D# packing per CDNA5 ISA 8.3/8.4 (group0 128b, group1 256b, type=2).
__device__ __forceinline__ void tdm_load_tile(const bf16* gsrc, unsigned lds_addr,
                                              int rows_rem, int cols_rem, int ld) {
    unsigned long long ga = (unsigned long long)(size_t)(const void*)gsrc;
    v4u g0;
    g0.x = 1u;                                            // count=1 (valid user D#)
    g0.y = lds_addr;                                      // LDS byte address
    g0.z = (unsigned)(ga & 0xFFFFFFFFu);                  // global_addr[31:0]
    g0.w = (unsigned)((ga >> 32) & 0x01FFFFFFu) | (2u << 30);  // addr[56:32] | type=2
    unsigned td0 = (unsigned)(cols_rem > 0 ? cols_rem : 0);    // tensor_dim0
    unsigned td1 = (unsigned)(rows_rem > 0 ? rows_rem : 0);    // tensor_dim1
    unsigned long long st0 = (unsigned long long)(unsigned)ld; // dim0 stride (elems)
    v8i g1;
    g1[0] = (int)((1u << 16)        // data_size = 2 bytes
                | (1u << 20)        // pad_enable
                | (4u << 22)        // pad_interval: 32 DWORDs (128B row)
                | (3u << 25));      // pad_amount: 4 DWORDs (16B)
    g1[1] = (int)((td0 & 0xFFFFu) << 16);                       // tensor_dim0[15:0]
    g1[2] = (int)(((td0 >> 16) & 0xFFFFu) | ((td1 & 0xFFFFu) << 16));
    g1[3] = (int)(((td1 >> 16) & 0xFFFFu) | (64u << 16));       // tile_dim0 = 64
    g1[4] = (int)(128u);                                        // tile_dim1 = 128
    g1[5] = (int)(st0 & 0xFFFFFFFFu);
    g1[6] = (int)((st0 >> 32) & 0xFFFFu);
    g1[7] = 0;
    v4i z4 = {0, 0, 0, 0};
#if __clang_major__ >= 23
    v8i z8 = {0, 0, 0, 0, 0, 0, 0, 0};
    __builtin_amdgcn_tensor_load_to_lds(g0, g1, z4, z4, z8, 0);
#else
    __builtin_amdgcn_tensor_load_to_lds(g0, g1, z4, z4, 0);
#endif
}
#endif

// ---------------------------------------------------------------------------
// Tiled bf16 WMMA GEMM:  C = act(alpha * A*B^T + bias) + R
//   A: M x K row-major (bf16); B: N x K row-major (weights pre-transposed)
//   C: fp32 (Cf) and/or bf16 (Cb), ld ldc; R optional residual (ldc)
//   batched over gridDim.z: bz -> (b = bz/nh, h = bz%nh), per-operand strides.
// Workgroup 256 thr = 8 waves; block tile 128x128x64, double-buffered LDS.
// Staging: TDM tensor_load_to_lds (TENSORcnt) when available, else
// register-prefetched global_load_b128 -> ds_store_b128.
// Each wave: 64 (M) x 32 (N) -> 16 v_wmma_f32_16x16x32_bf16 per K-stage.
// ---------------------------------------------------------------------------
__global__ __launch_bounds__(256) void k_gemm(
    const bf16* __restrict__ A, const bf16* __restrict__ B,
    const float* __restrict__ bias, const float* __restrict__ R,
    float* __restrict__ Cf, bf16* __restrict__ Cb,
    int M, int N, int K, int lda, int ldb, int ldc,
    long sAb, long sAh, long sBb, long sBh, long sCb, long sCh, int nh,
    float alpha, int flags)
{
    __shared__ __align__(16) bf16 sA[2][128][72];
    __shared__ __align__(16) bf16 sB[2][128][72];

    const int tid   = threadIdx.x;
    const int lane  = tid & 31;
    const int wid   = tid >> 5;
    const int waveM = wid >> 2;      // 0..1
    const int waveN = wid & 3;       // 0..3

    const int bz = blockIdx.z;
    const int bb = bz / nh;
    const int hh = bz - bb * nh;
    A += (long)bb * sAb + (long)hh * sAh;
    B += (long)bb * sBb + (long)hh * sBh;
    const long coff = (long)bb * sCb + (long)hh * sCh;
    if (Cf) Cf += coff;
    if (Cb) Cb += coff;
    if (R)  R  += coff;

    const int m0 = blockIdx.y * 128;
    const int n0 = blockIdx.x * 128;

    v8f acc[4][2];
#pragma unroll
    for (int i = 0; i < 4; i++)
#pragma unroll
        for (int j = 0; j < 2; j++)
#pragma unroll
            for (int r = 0; r < 8; r++) acc[i][j][r] = 0.0f;

    const int lm   = lane & 15;
    const int hsel = lane >> 4;
    const int KT   = (K + 63) >> 6;

#if !GEMM_TDM
    const int r2   = tid >> 1;          // 0..127 : tile row (A) / tile row (B^T)
    const int koff = (tid & 1) * 32;    // 0 or 32
    const bool aok = (m0 + r2) < M;
    const bool bok = (n0 + r2) < N;
    const bf16* arow = A + (size_t)(m0 + r2) * lda;
    const bf16* brow = B + (size_t)(n0 + r2) * ldb;
    v8bf ar[4], br[4];
    load_chunk32(arow + koff, aok, K - koff, ar);
    load_chunk32(brow + koff, bok, K - koff, br);
#pragma unroll
    for (int q = 0; q < 4; q++) *(v8bf*)&sA[0][r2][koff + q * 8] = ar[q];
#pragma unroll
    for (int q = 0; q < 4; q++) *(v8bf*)&sB[0][r2][koff + q * 8] = br[q];
    __syncthreads();
#else
    // prologue: DMA tile 0 (one issue per workgroup; EXEC!=0 only in wave 0)
    if (tid == 0) {
        tdm_load_tile(A + (size_t)m0 * lda, (unsigned)(size_t)(void*)&sA[0][0][0],
                      M - m0, K, lda);
        tdm_load_tile(B + (size_t)n0 * ldb, (unsigned)(size_t)(void*)&sB[0][0][0],
                      N - n0, K, ldb);
    }
#endif

    int buf = 0;
    for (int kt = 0; kt < KT; ++kt) {
        const bool more = (kt + 1) < KT;
#if GEMM_TDM
        if (tid == 0) __builtin_amdgcn_s_wait_tensorcnt((short)0);
        __syncthreads();                 // tile `buf` visible to all waves
        if (more && tid == 0) {          // DMA next tile while computing
            const int kb = (kt + 1) << 6;
            tdm_load_tile(A + (size_t)m0 * lda + kb,
                          (unsigned)(size_t)(void*)&sA[buf ^ 1][0][0],
                          M - m0, K - kb, lda);
            tdm_load_tile(B + (size_t)n0 * ldb + kb,
                          (unsigned)(size_t)(void*)&sB[buf ^ 1][0][0],
                          N - n0, K - kb, ldb);
        }
#else
        if (more) {                      // issue next tile's global loads
            const int kb = ((kt + 1) << 6) + koff;
            load_chunk32(arow + kb, aok, K - kb, ar);
            load_chunk32(brow + kb, bok, K - kb, br);
        }
#endif
        // 2 K-halves x (4 M-tiles x 2 N-tiles) WMMAs from LDS buffer `buf`
#pragma unroll
        for (int ks = 0; ks < 2; ks++) {
            v16bf af[4], bfr[2];
#pragma unroll
            for (int mt = 0; mt < 4; mt++) {
                const bf16* rp = &sA[buf][waveM * 64 + mt * 16 + lm][ks * 32];
                v8bf lo = *(const v8bf*)(rp + hsel * 8);
                v8bf hi = *(const v8bf*)(rp + 16 + hsel * 8);
                af[mt] = __builtin_shufflevector(lo, hi, 0,1,2,3,4,5,6,7,8,9,10,11,12,13,14,15);
            }
#pragma unroll
            for (int nt = 0; nt < 2; nt++) {
                const bf16* rp = &sB[buf][waveN * 32 + nt * 16 + lm][ks * 32];
                v8bf lo = *(const v8bf*)(rp + hsel * 8);
                v8bf hi = *(const v8bf*)(rp + 16 + hsel * 8);
                bfr[nt] = __builtin_shufflevector(lo, hi, 0,1,2,3,4,5,6,7,8,9,10,11,12,13,14,15);
            }
#pragma unroll
            for (int mt = 0; mt < 4; mt++)
#pragma unroll
                for (int nt = 0; nt < 2; nt++)
                    acc[mt][nt] = __builtin_amdgcn_wmma_f32_16x16x32_bf16(
                        false, af[mt], false, bfr[nt], (short)0, acc[mt][nt], false, false);
        }
#if !GEMM_TDM
        if (more) {                      // store prefetched tile into other buffer
#pragma unroll
            for (int q = 0; q < 4; q++) *(v8bf*)&sA[buf ^ 1][r2][koff + q * 8] = ar[q];
#pragma unroll
            for (int q = 0; q < 4; q++) *(v8bf*)&sB[buf ^ 1][r2][koff + q * 8] = br[q];
        }
#endif
        __syncthreads();                 // readers done before `buf` is overwritten
        buf ^= 1;
    }

    // ---- epilogue: C layout = VGPR r -> row r (lanes 0-15) / r+8 (16-31) ----
#pragma unroll
    for (int mt = 0; mt < 4; mt++) {
#pragma unroll
        for (int nt = 0; nt < 2; nt++) {
#pragma unroll
            for (int r = 0; r < 8; r++) {
                const int grow = m0 + waveM * 64 + mt * 16 + r + hsel * 8;
                const int gcol = n0 + waveN * 32 + nt * 16 + lm;
                if (grow < M && gcol < N) {
                    float v = acc[mt][nt][r] * alpha;
                    if (bias)      v += bias[gcol];
                    if (flags & 2) v  = gelu_exact(v);
                    if (R)         v += R[(size_t)grow * ldc + gcol];
                    if (Cf) Cf[(size_t)grow * ldc + gcol] = v;
                    if (Cb) Cb[(size_t)grow * ldc + gcol] = (bf16)v;
                }
            }
        }
    }
}

// ---------------- elementwise / reduction kernels ----------------

__global__ void k_f2b(const float* __restrict__ in, bf16* __restrict__ out, long n) {
    long i = (long)blockIdx.x * 256 + threadIdx.x;
    if (i < n) out[i] = (bf16)in[i];
}

// fp32 [L][K][N] -> bf16 [L][N][K]  (weights pre-transposed for NT GEMM)
__global__ void k_f2bT(const float* __restrict__ in, bf16* __restrict__ out,
                       int K, int N, long total) {
    long i = (long)blockIdx.x * 256 + threadIdx.x;
    if (i >= total) return;
    int  k = (int)(i % K);
    long r = i / K;
    int  n = (int)(r % N);
    int  l = (int)(r / N);
    out[i] = (bf16)in[((long)l * K + k) * N + n];
}

// im2col for 16x16/stride-16 patches, NCHW -> (B*196, 768) bf16
__global__ void k_im2col(const float* __restrict__ x, bf16* __restrict__ A) {
    long i = (long)blockIdx.x * 256 + threadIdx.x;
    long total = (long)NBATCH * NPATCH * DIMV;
    if (i >= total) return;
    int  e = (int)(i % DIMV);
    long r = i / DIMV;
    int  t = (int)(r % NPATCH);
    int  b = (int)(r / NPATCH);
    int  c   = e >> 8;
    int  rem = e & 255;
    int  ii = rem >> 4, jj = rem & 15;
    int  ph = t / 14,  pw = t % 14;
    A[i] = (bf16)x[(((long)b * 3 + c) * 224 + ph * 16 + ii) * 224 + pw * 16 + jj];
}

// z[b,t,:] = (t==0 ? cls : patchC[b,t-1,:]) + pos[t,:]
__global__ void k_build_z(const float* __restrict__ pc, const float* __restrict__ cls,
                          const float* __restrict__ pos, float* __restrict__ z) {
    long i = (long)blockIdx.x * 256 + threadIdx.x;
    long total = (long)NBATCH * NTOK1 * DIMV;
    if (i >= total) return;
    int  d = (int)(i % DIMV);
    long r = i / DIMV;
    int  t = (int)(r % NTOK1);
    int  b = (int)(r / NTOK1);
    float v = (t == 0) ? cls[d] : pc[((long)b * NPATCH + (t - 1)) * DIMV + d];
    z[i] = v + pos[(long)t * DIMV + d];
}

// V slice of qkv (bf16, token-major) -> Vt[bh][d][t] with leading dim SLD
__global__ void k_transpose_v(const bf16* __restrict__ qkvb, bf16* __restrict__ Vt, int ntok) {
    long i = (long)blockIdx.x * 256 + threadIdx.x;
    long total = (long)NBATCH * HEADS * HDIM * ntok;
    if (i >= total) return;
    int  t  = (int)(i % ntok);
    long r  = i / ntok;
    int  d  = (int)(r % HDIM);
    int  bh = (int)(r / HDIM);
    int  b  = bh / HEADS, h = bh % HEADS;
    Vt[((long)bh * HDIM + d) * SLD + t] =
        qkvb[((long)b * ntok + t) * (3 * DIMV) + 2 * DIMV + (long)h * HDIM + d];
}

// LayerNorm: one block per row, fp32 in -> bf16 out
__global__ __launch_bounds__(256) void k_ln(const float* __restrict__ x,
                                            const float* __restrict__ g,
                                            const float* __restrict__ bparm,
                                            bf16* __restrict__ out, int dim) {
    __shared__ float red[256];
    const long row = blockIdx.x;
    const float* xr = x + row * (long)dim;
    float s = 0.f, s2 = 0.f;
    for (int d = threadIdx.x; d < dim; d += 256) { float v = xr[d]; s += v; s2 += v * v; }
    red[threadIdx.x] = s; __syncthreads();
    for (int o = 128; o > 0; o >>= 1) { if (threadIdx.x < o) red[threadIdx.x] += red[threadIdx.x + o]; __syncthreads(); }
    const float mean = red[0] / dim;
    __syncthreads();
    red[threadIdx.x] = s2; __syncthreads();
    for (int o = 128; o > 0; o >>= 1) { if (threadIdx.x < o) red[threadIdx.x] += red[threadIdx.x + o]; __syncthreads(); }
    const float var = red[0] / dim - mean * mean;
    const float inv = rsqrtf(var + 1e-6f);
    for (int d = threadIdx.x; d < dim; d += 256)
        out[row * (long)dim + d] = (bf16)((xr[d] - mean) * inv * g[d] + bparm[d]);
}

// row softmax over n entries (leading dim SLD); fp32 in-place + bf16 copy
__global__ __launch_bounds__(256) void k_softmax(float* __restrict__ S, bf16* __restrict__ P, int n) {
    __shared__ float red[256];
    const long gid = blockIdx.x;
    const long bh  = gid / n;
    const int  r   = (int)(gid % n);
    float* row = S + bh * (long)(SLD * SLD) + (long)r * SLD;
    bf16*  prw = P + bh * (long)(SLD * SLD) + (long)r * SLD;
    float mx = -3.0e38f;
    for (int j = threadIdx.x; j < n; j += 256) mx = fmaxf(mx, row[j]);
    red[threadIdx.x] = mx; __syncthreads();
    for (int o = 128; o > 0; o >>= 1) { if (threadIdx.x < o) red[threadIdx.x] = fmaxf(red[threadIdx.x], red[threadIdx.x + o]); __syncthreads(); }
    mx = red[0]; __syncthreads();
    float s = 0.f;
    for (int j = threadIdx.x; j < n; j += 256) { float e = __expf(row[j] - mx); row[j] = e; s += e; }
    red[threadIdx.x] = s; __syncthreads();
    for (int o = 128; o > 0; o >>= 1) { if (threadIdx.x < o) red[threadIdx.x] += red[threadIdx.x + o]; __syncthreads(); }
    const float inv = 1.0f / red[0];
    for (int j = threadIdx.x; j < n; j += 256) { float p = row[j] * inv; row[j] = p; prw[j] = (bf16)p; }
}

// per-patch-token L2 norm of z (pre-prune layout)
__global__ __launch_bounds__(256) void k_norms(const float* __restrict__ z, float* __restrict__ norms) {
    __shared__ float red[256];
    const int b = blockIdx.x / NPATCH;
    const int t = blockIdx.x % NPATCH;
    const float* xr = z + ((long)b * NTOK1 + 1 + t) * DIMV;
    float s = 0.f;
    for (int d = threadIdx.x; d < DIMV; d += 256) { float v = xr[d]; s += v * v; }
    red[threadIdx.x] = s; __syncthreads();
    for (int o = 128; o > 0; o >>= 1) { if (threadIdx.x < o) red[threadIdx.x] += red[threadIdx.x + o]; __syncthreads(); }
    if (threadIdx.x == 0) norms[blockIdx.x] = sqrtf(red[0]);
}

// top-KEEPT by norm (ties -> lower index, matching lax.top_k), index-sorted gather
__global__ __launch_bounds__(256) void k_prune(const float* __restrict__ z,
                                               const float* __restrict__ norms,
                                               float* __restrict__ z2) {
    __shared__ float sn[NPATCH];
    __shared__ int   keepf[NPATCH];
    __shared__ int   map[NTOK2];
    const int b = blockIdx.x, tid = threadIdx.x;
    if (tid < NPATCH) sn[tid] = norms[b * NPATCH + tid];
    __syncthreads();
    if (tid < NPATCH) {
        float mine = sn[tid]; int rank = 0;
        for (int j = 0; j < NPATCH; j++) {
            float nj = sn[j];
            rank += (nj > mine) || (nj == mine && j < tid);
        }
        keepf[tid] = (rank < KEEPT) ? 1 : 0;
    }
    __syncthreads();
    if (tid == 0) map[0] = 0;
    if (tid < NPATCH && keepf[tid]) {
        int pos = 0;
        for (int j = 0; j < tid; j++) pos += keepf[j];
        map[1 + pos] = 1 + tid;
    }
    __syncthreads();
    for (long i = tid; i < (long)NTOK2 * DIMV; i += 256) {
        int r = (int)(i / DIMV), d = (int)(i % DIMV);
        z2[((long)b * NTOK2 + r) * DIMV + d] = z[((long)b * NTOK1 + map[r]) * DIMV + d];
    }
}

__global__ void k_gather_cls(const float* __restrict__ z, float* __restrict__ out, int ntok) {
    const int b = blockIdx.x;
    for (int d = threadIdx.x; d < DIMV; d += 256)
        out[(long)b * DIMV + d] = z[(long)b * ntok * DIMV + d];
}

// ---------------------------------------------------------------------------
extern "C" void kernel_launch(void* const* d_in, const int* in_sizes, int n_in,
                              void* d_out, int out_size, void* d_ws, size_t ws_size,
                              hipStream_t stream) {
    (void)in_sizes; (void)n_in; (void)out_size; (void)ws_size;
    const float* x         = (const float*)d_in[0];
    const float* patch_w   = (const float*)d_in[1];
    const float* patch_b   = (const float*)d_in[2];
    const float* cls_token = (const float*)d_in[3];
    const float* pos_embed = (const float*)d_in[4];
    const float* ln1_g     = (const float*)d_in[5];
    const float* ln1_b     = (const float*)d_in[6];
    const float* qkv_w     = (const float*)d_in[7];
    const float* qkv_b     = (const float*)d_in[8];
    const float* proj_w    = (const float*)d_in[9];
    const float* proj_b    = (const float*)d_in[10];
    const float* ln2_g     = (const float*)d_in[11];
    const float* ln2_b     = (const float*)d_in[12];
    const float* fc1_w     = (const float*)d_in[13];
    const float* fc1_b     = (const float*)d_in[14];
    const float* fc2_w     = (const float*)d_in[15];
    const float* fc2_b     = (const float*)d_in[16];
    const float* norm_g    = (const float*)d_in[17];
    const float* norm_b    = (const float*)d_in[18];
    const float* head_w    = (const float*)d_in[19];
    const float* head_b    = (const float*)d_in[20];

    char* wp = (char*)d_ws;
    auto alloc = [&](size_t bytes) -> char* {
        char* p = wp; wp += (bytes + 255) & ~(size_t)255; return p;
    };
    auto cdiv = [](int a, int b) { return (a + b - 1) / b; };

    bf16*  wb_patch = (bf16*)alloc((size_t)DIMV * DIMV * 2);                 // already NxK
    bf16*  wb_qkvT  = (bf16*)alloc((size_t)DEPTH * 3 * DIMV * DIMV * 2);     // [l][2304][768]
    bf16*  wb_projT = (bf16*)alloc((size_t)DEPTH * DIMV * DIMV * 2);         // [l][768][768]
    bf16*  wb_fc1T  = (bf16*)alloc((size_t)DEPTH * MLPD * DIMV * 2);         // [l][3072][768]
    bf16*  wb_fc2T  = (bf16*)alloc((size_t)DEPTH * DIMV * MLPD * 2);         // [l][768][3072]
    bf16*  wb_headT = (bf16*)alloc((size_t)NCLS * DIMV * 2);                 // [1000][768]
    bf16*  Aim      = (bf16*)alloc((size_t)NBATCH * NPATCH * DIMV * 2);
    float* patchC   = (float*)alloc((size_t)NBATCH * NPATCH * DIMV * 4);
    float* z        = (float*)alloc((size_t)NBATCH * NTOK1 * DIMV * 4);
    float* z2       = (float*)alloc((size_t)NBATCH * NTOK2 * DIMV * 4);
    bf16*  hbf      = (bf16*)alloc((size_t)NBATCH * NTOK1 * DIMV * 2);
    bf16*  qkvb     = (bf16*)alloc((size_t)NBATCH * NTOK1 * 3 * DIMV * 2);
    bf16*  Vt       = (bf16*)alloc((size_t)NBATCH * HEADS * HDIM * SLD * 2);
    float* S        = (float*)alloc((size_t)NBATCH * HEADS * SLD * SLD * 4);
    bf16*  P        = (bf16*)alloc((size_t)NBATCH * HEADS * SLD * SLD * 2);
    bf16*  Obf      = (bf16*)alloc((size_t)NBATCH * NTOK1 * DIMV * 2);
    bf16*  Mbf      = (bf16*)alloc((size_t)NBATCH * NTOK1 * MLPD * 2);
    float* norms    = (float*)alloc((size_t)NBATCH * NPATCH * 4);
    float* cls32    = (float*)alloc((size_t)NBATCH * DIMV * 4);
    bf16*  clsbf    = (bf16*)alloc((size_t)NBATCH * DIMV * 2);

    const dim3 blk(256);

    // ---- one-shot weight conversion fp32 -> bf16 (transposed to NxK) ----
    {
        long n = (long)DIMV * DIMV;  // patch_w is [out][in] == NxK already
        k_f2b<<<(int)((n + 255) / 256), blk, 0, stream>>>(patch_w, wb_patch, n);
        long t;
        t = (long)DEPTH * 3 * DIMV * DIMV;
        k_f2bT<<<(int)((t + 255) / 256), blk, 0, stream>>>(qkv_w,  wb_qkvT,  DIMV, 3 * DIMV, t);
        t = (long)DEPTH * DIMV * DIMV;
        k_f2bT<<<(int)((t + 255) / 256), blk, 0, stream>>>(proj_w, wb_projT, DIMV, DIMV, t);
        t = (long)DEPTH * MLPD * DIMV;
        k_f2bT<<<(int)((t + 255) / 256), blk, 0, stream>>>(fc1_w,  wb_fc1T,  DIMV, MLPD, t);
        t = (long)DEPTH * DIMV * MLPD;
        k_f2bT<<<(int)((t + 255) / 256), blk, 0, stream>>>(fc2_w,  wb_fc2T,  MLPD, DIMV, t);
        t = (long)NCLS * DIMV;
        k_f2bT<<<(int)((t + 255) / 256), blk, 0, stream>>>(head_w, wb_headT, DIMV, NCLS, t);
    }

    // ---- patch embedding: im2col + GEMM against patch_w (NxK) ----
    {
        long nim = (long)NBATCH * NPATCH * DIMV;
        k_im2col<<<(int)((nim + 255) / 256), blk, 0, stream>>>(x, Aim);
        const int Mp = NBATCH * NPATCH;
        k_gemm<<<dim3(cdiv(DIMV, 128), cdiv(Mp, 128), 1), blk, 0, stream>>>(
            Aim, wb_patch, patch_b, nullptr, patchC, nullptr,
            Mp, DIMV, DIMV, DIMV, DIMV, DIMV,
            0, 0, 0, 0, 0, 0, 1, 1.0f, 0);
        long nz = (long)NBATCH * NTOK1 * DIMV;
        k_build_z<<<(int)((nz + 255) / 256), blk, 0, stream>>>(patchC, cls_token, pos_embed, z);
    }

    float* zc  = z;
    int Ntok   = NTOK1;
    const float scale = 0.125f;   // 1/sqrt(64)

    for (int i = 0; i < DEPTH; i++) {
        const int T = NBATCH * Ntok;

        // LN1 -> bf16
        k_ln<<<T, blk, 0, stream>>>(zc, ln1_g + i * DIMV, ln1_b + i * DIMV, hbf, DIMV);

        // QKV: (T,768) x (2304,768)^T + bias -> bf16
        k_gemm<<<dim3(cdiv(3 * DIMV, 128), cdiv(T, 128), 1), blk, 0, stream>>>(
            hbf, wb_qkvT + (size_t)i * 3 * DIMV * DIMV, qkv_b + i * 3 * DIMV,
            nullptr, nullptr, qkvb,
            T, 3 * DIMV, DIMV, DIMV, DIMV, 3 * DIMV,
            0, 0, 0, 0, 0, 0, 1, 1.0f, 0);

        // V -> Vt[bh][d][t] (padded ld SLD) so P*V is an NT GEMM too
        {
            long nv = (long)NBATCH * HEADS * HDIM * Ntok;
            k_transpose_v<<<(int)((nv + 255) / 256), blk, 0, stream>>>(qkvb, Vt, Ntok);
        }

        // S = scale * Q K^T  (batched over b,h; K rows are tokens -> NT)
        k_gemm<<<dim3(cdiv(Ntok, 128), cdiv(Ntok, 128), NBATCH * HEADS), blk, 0, stream>>>(
            qkvb, qkvb + DIMV, nullptr, nullptr, S, nullptr,
            Ntok, Ntok, HDIM, 3 * DIMV, 3 * DIMV, SLD,
            (long)Ntok * 3 * DIMV, HDIM, (long)Ntok * 3 * DIMV, HDIM,
            (long)HEADS * SLD * SLD, (long)SLD * SLD,
            HEADS, scale, 0);

        // softmax rows -> P (bf16)
        k_softmax<<<NBATCH * HEADS * Ntok, blk, 0, stream>>>(S, P, Ntok);

        // O = P V  (B = Vt, N=64 x K=Ntok), scattered into (T,768) at col h*64
        k_gemm<<<dim3(1, cdiv(Ntok, 128), NBATCH * HEADS), blk, 0, stream>>>(
            P, Vt, nullptr, nullptr, nullptr, Obf,
            Ntok, HDIM, Ntok, SLD, SLD, DIMV,
            (long)HEADS * SLD * SLD, (long)SLD * SLD,
            (long)HEADS * HDIM * SLD, (long)HDIM * SLD,
            (long)Ntok * DIMV, HDIM,
            HEADS, 1.0f, 0);

        // proj + bias + residual (in-place fp32 z)
        k_gemm<<<dim3(cdiv(DIMV, 128), cdiv(T, 128), 1), blk, 0, stream>>>(
            Obf, wb_projT + (size_t)i * DIMV * DIMV, proj_b + i * DIMV,
            zc, zc, nullptr,
            T, DIMV, DIMV, DIMV, DIMV, DIMV,
            0, 0, 0, 0, 0, 0, 1, 1.0f, 0);

        // LN2 -> bf16
        k_ln<<<T, blk, 0, stream>>>(zc, ln2_g + i * DIMV, ln2_b + i * DIMV, hbf, DIMV);

        // fc1 + bias + exact GELU -> bf16
        k_gemm<<<dim3(cdiv(MLPD, 128), cdiv(T, 128), 1), blk, 0, stream>>>(
            hbf, wb_fc1T + (size_t)i * MLPD * DIMV, fc1_b + i * MLPD,
            nullptr, nullptr, Mbf,
            T, MLPD, DIMV, DIMV, DIMV, MLPD,
            0, 0, 0, 0, 0, 0, 1, 1.0f, /*gelu*/2);

        // fc2 + bias + residual (in-place fp32 z)
        k_gemm<<<dim3(cdiv(DIMV, 128), cdiv(T, 128), 1), blk, 0, stream>>>(
            Mbf, wb_fc2T + (size_t)i * DIMV * MLPD, fc2_b + i * DIMV,
            zc, zc, nullptr,
            T, DIMV, MLPD, MLPD, MLPD, DIMV,
            0, 0, 0, 0, 0, 0, 1, 1.0f, 0);

        // token pruning after layer 6
        if (i == 5) {
            k_norms<<<NBATCH * NPATCH, blk, 0, stream>>>(zc, norms);
            k_prune<<<NBATCH, blk, 0, stream>>>(zc, norms, z2);
            zc = z2; Ntok = NTOK2;
        }
    }

    // ---- final LN on cls tokens + classification head ----
    k_gather_cls<<<NBATCH, blk, 0, stream>>>(zc, cls32, Ntok);
    k_ln<<<NBATCH, blk, 0, stream>>>(cls32, norm_g, norm_b, clsbf, DIMV);
    k_gemm<<<dim3(cdiv(NCLS, 128), 1, 1), blk, 0, stream>>>(
        clsbf, wb_headT, head_b, nullptr, (float*)d_out, nullptr,
        NBATCH, NCLS, DIMV, DIMV, DIMV, NCLS,
        0, 0, 0, 0, 0, 0, 1, 1.0f, 0);
}
